// MPNN_22230750724232
// MI455X (gfx1250) — matrix operand order repaired and checked by reading, not compiled, not run on hardware
//
#include <hip/hip_runtime.h>

// Problem sizes (match reference)
#define NN 50000
#define EE 500000
#define GG 1024
#define NDI 32
#define EDI 16
#define HH 128
#define LL 4

#if defined(__has_builtin)
#if __has_builtin(__builtin_amdgcn_global_load_async_to_lds_b128)
#define USE_ASYNC_LDS 1
#endif
#endif

typedef __attribute__((ext_vector_type(8)))  __bf16 bf16x8;
typedef __attribute__((ext_vector_type(16))) __bf16 bf16x16;
typedef __attribute__((ext_vector_type(8)))  float  f32x8;

// GCC-style int4 vector: exact parameter type of the async-LDS builtins.
typedef int v4i __attribute__((vector_size(16)));
typedef __attribute__((address_space(1))) v4i* gv4i_ptr;
typedef __attribute__((address_space(3))) v4i* sv4i_ptr;

static __device__ __forceinline__ bf16x16 combine8(bf16x8 lo, bf16x8 hi) {
  bf16x16 r;
#pragma unroll
  for (int i = 0; i < 8; ++i) { r[i] = lo[i]; r[i + 8] = hi[i]; }
  return r;
}
// A-fragment load: per-lane 8 contiguous bf16 at p, 8 contiguous at p+16
// (ISA 16-bit A 16x32 layout: lanes0-15 K={0..7,16..23}, lanes16-31 K={8..15,24..31};
//  caller adds hsel*8 to p)
static __device__ __forceinline__ bf16x16 ldA(const __bf16* p) {
  return combine8(*(const bf16x8*)p, *(const bf16x8*)(p + 16));
}
// B-fragment load: per-lane 8 contiguous bf16 at p, 8 contiguous at p+16 (weights stored column-major)
static __device__ __forceinline__ bf16x16 ldB(const __bf16* p) {
  return combine8(*(const bf16x8*)p, *(const bf16x8*)(p + 16));
}

// Hardware float atomic (global_atomic_add_f32) — avoids CAS expansion.
static __device__ __forceinline__ void atomicAddF32(float* p, float v) {
  unsafeAtomicAdd(p, v);
}

// ---------------------------------------------------------------- weight prep
// dst[n*K + k] = (bf16) src[k*Nc + n]   (transpose to column-major bf16)
__global__ void transpose_cvt(const float* __restrict__ src, __bf16* __restrict__ dst,
                              int K, int Nc) {
  int idx = blockIdx.x * blockDim.x + threadIdx.x;
  if (idx >= K * Nc) return;
  int k = idx / Nc, n = idx - k * Nc;
  dst[n * K + k] = (__bf16)src[idx];
}

// ---------------------------------------------------------------- embeddings
__global__ void node_embed(const float* __restrict__ x, const float* __restrict__ w,
                           const float* __restrict__ b, float* __restrict__ h,
                           __bf16* __restrict__ h_bf, float* __restrict__ agg) {
  int n = blockIdx.x, c = threadIdx.x;
  __shared__ float sx[NDI];
  if (c < NDI) sx[c] = x[(size_t)n * NDI + c];
  __syncthreads();
  float acc = b[c];
#pragma unroll
  for (int k = 0; k < NDI; ++k) acc = fmaf(sx[k], w[k * HH + c], acc);
  size_t o = (size_t)n * HH + c;
  h[o] = acc; h_bf[o] = (__bf16)acc; agg[o] = 0.f;
}

__global__ void edge_embed(const float* __restrict__ ea, const float* __restrict__ w,
                           const float* __restrict__ b, __bf16* __restrict__ e_bf) {
  int e = blockIdx.x, c = threadIdx.x;
  __shared__ float sx[EDI];
  if (c < EDI) sx[c] = ea[(size_t)e * EDI + c];
  __syncthreads();
  float acc = b[c];
#pragma unroll
  for (int k = 0; k < EDI; ++k) acc = fmaf(sx[k], w[k * HH + c], acc);
  e_bf[(size_t)e * HH + c] = (__bf16)acc;
}

// ---------------------------------------------------------------- fused layer
// Per layer: e = relu(e@W1+b1)@W2+b2 ; msg = BN(relu([h[src]|e]@NW1+nb1)@NW2+nb2)
//            agg[dst] += msg  (atomics). Weights resident in LDS, bf16 WMMA.
#define WBLK_ELEMS 81920          // em1t(16384) em2t(16384) nm1t(32768) nm2t(16384)
#define STAGE_ELEMS (16 * 256)    // per-wave staging tile (row-major, ld=256)

__global__ void __launch_bounds__(256, 1)
fused_edge_layer(const int* __restrict__ srcI, const int* __restrict__ dstI,
                 const __bf16* __restrict__ h_bf, __bf16* __restrict__ e_bf,
                 float* __restrict__ agg, const __bf16* __restrict__ wblk,
                 const float* __restrict__ eb1, const float* __restrict__ eb2,
                 const float* __restrict__ nb1, const float* __restrict__ nb2,
                 const float* __restrict__ bng, const float* __restrict__ bnb) {
  extern __shared__ char smem_raw[];
  __bf16* sW = (__bf16*)smem_raw;
  const int lane = threadIdx.x & 31;
  const int wave = threadIdx.x >> 5;
  __bf16* sStage = sW + WBLK_ELEMS + wave * STAGE_ELEMS;
  const int colL = lane & 15;
  const int hsel = lane >> 4;     // 0: lanes 0-15, 1: lanes 16-31
  const int hi8  = hsel * 8;
  const float inv_std = 0.9999950000374997f;   // 1/sqrt(1+1e-5), eval BN

  // Prefetch weight block into cache hierarchy (global_prefetch_b8), then
  // stage all four matrices (160 KB bf16) into LDS once per WG.
  for (int i = threadIdx.x * 64; i < WBLK_ELEMS; i += blockDim.x * 64)
    __builtin_prefetch(wblk + i, 0, 1);
  {
    const int n16 = WBLK_ELEMS * 2 / 16;   // 10240 x 16B chunks
#ifdef USE_ASYNC_LDS
    // Memory -> LDS direct (ASYNCcnt path): no VGPR round-trip.
    gv4i_ptr gp = (gv4i_ptr)wblk;
    sv4i_ptr sp = (sv4i_ptr)sW;
    for (int i = threadIdx.x; i < n16; i += blockDim.x)
      __builtin_amdgcn_global_load_async_to_lds_b128(gp + i, sp + i, 0, 0);
#if defined(__has_builtin) && __has_builtin(__builtin_amdgcn_s_wait_asynccnt)
    __builtin_amdgcn_s_wait_asynccnt(0);
#else
    asm volatile("s_wait_asynccnt 0x0" ::: "memory");
#endif
#else
    const uint4* g = (const uint4*)wblk;
    uint4* s = (uint4*)sW;
#pragma unroll 4
    for (int i = threadIdx.x; i < n16; i += blockDim.x) s[i] = g[i];
#endif
  }
  __syncthreads();

  const __bf16* W1  = sW;                    // [128][128] col-major (K=128)
  const __bf16* W2  = sW + 16384;            // [128][128]
  const __bf16* NW1 = sW + 32768;            // [128][256] col-major (K=256)
  const __bf16* NW2 = sW + 65536;            // [128][128]

  const int nWaveTiles = EE / 16;            // 31250, exact
  const int nWGTiles = (nWaveTiles + 7) / 8;
  for (int t = blockIdx.x; t < nWGTiles; t += gridDim.x) {
    const int wtile = t * 8 + wave;
    if (wtile >= nWaveTiles) continue;       // no barriers inside loop
    const int ebase = wtile * 16;

    bf16x16 a[4];
    // -------- Stage 1: t1 = relu(e @ W1 + b1) -> sStage cols [0,128)
    {
      const __bf16* eRow = e_bf + (size_t)(ebase + colL) * HH;
#pragma unroll
      for (int k = 0; k < 4; ++k) a[k] = ldA(eRow + k * 32 + hsel * 8);
    }
#pragma unroll
    for (int n = 0; n < 8; ++n) {
      f32x8 acc = {};
#pragma unroll
      for (int k = 0; k < 4; ++k) {
        bf16x16 b = ldB(W1 + (n * 16 + colL) * 128 + k * 32 + hsel * 8);
        acc = __builtin_amdgcn_wmma_f32_16x16x32_bf16(false, a[k], false, b,
                                                      (short)0, acc, false, false);
      }
      float bias = eb1[n * 16 + colL];
#pragma unroll
      for (int i = 0; i < 8; ++i) {
        float v = fmaxf(acc[i] + bias, 0.f);
        sStage[(hi8 + i) * 256 + n * 16 + colL] = (__bf16)v;
      }
    }

    // -------- Stage 2: e_new = t1 @ W2 + b2 -> global e_bf + sStage cols [128,256)
#pragma unroll
    for (int k = 0; k < 4; ++k) a[k] = ldA(sStage + colL * 256 + k * 32 + hsel * 8);
#pragma unroll
    for (int n = 0; n < 8; ++n) {
      f32x8 acc = {};
#pragma unroll
      for (int k = 0; k < 4; ++k) {
        bf16x16 b = ldB(W2 + (n * 16 + colL) * 128 + k * 32 + hsel * 8);
        acc = __builtin_amdgcn_wmma_f32_16x16x32_bf16(false, a[k], false, b,
                                                      (short)0, acc, false, false);
      }
      float bias = eb2[n * 16 + colL];
#pragma unroll
      for (int i = 0; i < 8; ++i) {
        __bf16 bv = (__bf16)(acc[i] + bias);
        sStage[(hi8 + i) * 256 + 128 + n * 16 + colL] = bv;
        e_bf[(size_t)(ebase + hi8 + i) * HH + n * 16 + colL] = bv;
      }
    }

    // -------- Stage 3: t2 = relu([h[src] | e_new] @ NW1 + nb1) -> sStage cols [0,128)
    bf16x16 am[8];
    {
      int sid = srcI[ebase + colL];
      const __bf16* hRow = h_bf + (size_t)sid * HH;   // per-edge gather
#pragma unroll
      for (int k = 0; k < 4; ++k) am[k] = ldA(hRow + k * 32 + hsel * 8);
#pragma unroll
      for (int k = 0; k < 4; ++k)
        am[4 + k] = ldA(sStage + colL * 256 + 128 + k * 32 + hsel * 8);
    }
#pragma unroll
    for (int n = 0; n < 8; ++n) {
      f32x8 acc = {};
#pragma unroll
      for (int k = 0; k < 8; ++k) {
        bf16x16 b = ldB(NW1 + (n * 16 + colL) * 256 + k * 32 + hsel * 8);
        acc = __builtin_amdgcn_wmma_f32_16x16x32_bf16(false, am[k], false, b,
                                                      (short)0, acc, false, false);
      }
      float bias = nb1[n * 16 + colL];
#pragma unroll
      for (int i = 0; i < 8; ++i) {
        float v = fmaxf(acc[i] + bias, 0.f);
        sStage[(hi8 + i) * 256 + n * 16 + colL] = (__bf16)v;
      }
    }

    // -------- Stage 4: msg = BN(t2 @ NW2 + nb2); scatter-add to agg[dst]
#pragma unroll
    for (int k = 0; k < 4; ++k) a[k] = ldA(sStage + colL * 256 + k * 32 + hsel * 8);
    int drow[8];
#pragma unroll
    for (int i = 0; i < 8; ++i) drow[i] = dstI[ebase + hi8 + i];
#pragma unroll
    for (int n = 0; n < 8; ++n) {
      f32x8 acc = {};
#pragma unroll
      for (int k = 0; k < 4; ++k) {
        bf16x16 b = ldB(NW2 + (n * 16 + colL) * 128 + k * 32 + hsel * 8);
        acc = __builtin_amdgcn_wmma_f32_16x16x32_bf16(false, a[k], false, b,
                                                      (short)0, acc, false, false);
      }
      int c = n * 16 + colL;
      float sc = inv_std * bng[c];
      float sh = bnb[c];
      float bias = nb2[c];
#pragma unroll
      for (int i = 0; i < 8; ++i) {
        float v = fmaf(acc[i] + bias, sc, sh);
        atomicAddF32(agg + (size_t)drow[i] * HH + c, v);
      }
    }
  }
}

// ---------------------------------------------------------------- epilogue
__global__ void update_h(float* __restrict__ h, float* __restrict__ agg,
                         __bf16* __restrict__ h_bf, int total) {
  int i = blockIdx.x * blockDim.x + threadIdx.x;
  if (i >= total) return;
  float v = h[i] + agg[i];
  h[i] = v; h_bf[i] = (__bf16)v; agg[i] = 0.f;   // re-zero for next layer
}

__global__ void zero_f32(float* __restrict__ p, int n) {
  int i = blockIdx.x * blockDim.x + threadIdx.x;
  if (i < n) p[i] = 0.f;
}

__global__ void pool_scatter(const float* __restrict__ h, const int* __restrict__ batch,
                             float* __restrict__ pooled, int total) {
  int i = blockIdx.x * blockDim.x + threadIdx.x;
  if (i >= total) return;
  int n = i >> 7, c = i & 127;
  atomicAddF32(pooled + (size_t)batch[n] * HH + c, h[i]);
}

__global__ void readout(const float* __restrict__ pooled, const float* __restrict__ w1,
                        const float* __restrict__ b1, const float* __restrict__ w2,
                        const float* __restrict__ b2, float* __restrict__ out) {
  int g = blockIdx.x, c = threadIdx.x;   // 128 threads
  __shared__ float sp[HH];
  __shared__ float sr[HH];
  sp[c] = pooled[(size_t)g * HH + c];
  __syncthreads();
  float acc = b1[c];
#pragma unroll 8
  for (int k = 0; k < HH; ++k) acc = fmaf(sp[k], w1[k * HH + c], acc);
  sr[c] = fmaxf(acc, 0.f) * w2[c];
  __syncthreads();
#pragma unroll
  for (int s = 64; s > 0; s >>= 1) {
    if (c < s) sr[c] += sr[c + s];
    __syncthreads();
  }
  if (c == 0) out[g] = sr[0] + b2[0];
}

// ---------------------------------------------------------------- launcher
extern "C" void kernel_launch(void* const* d_in, const int* in_sizes, int n_in,
                              void* d_out, int out_size, void* d_ws, size_t ws_size,
                              hipStream_t stream) {
  (void)in_sizes; (void)n_in; (void)out_size; (void)ws_size;
  const float* x     = (const float*)d_in[0];
  const int*   ei    = (const int*)d_in[1];
  const float* ea    = (const float*)d_in[2];
  const int*   batch = (const int*)d_in[3];
  const float* ne_w  = (const float*)d_in[4];
  const float* ne_b  = (const float*)d_in[5];
  const float* ee_w  = (const float*)d_in[6];
  const float* ee_b  = (const float*)d_in[7];
  const float* nm_w1 = (const float*)d_in[8];
  const float* nm_b1 = (const float*)d_in[9];
  const float* nm_w2 = (const float*)d_in[10];
  const float* nm_b2 = (const float*)d_in[11];
  const float* bn_g  = (const float*)d_in[12];
  const float* bn_b  = (const float*)d_in[13];
  const float* em_w1 = (const float*)d_in[14];
  const float* em_b1 = (const float*)d_in[15];
  const float* em_w2 = (const float*)d_in[16];
  const float* em_b2 = (const float*)d_in[17];
  const float* ro_w1 = (const float*)d_in[18];
  const float* ro_b1 = (const float*)d_in[19];
  const float* ro_w2 = (const float*)d_in[20];
  const float* ro_b2 = (const float*)d_in[21];
  const int* srcI = ei;
  const int* dstI = ei + EE;

  char* p = (char*)d_ws;
  auto alloc = [&](size_t bytes) {
    char* r = p; p += (bytes + 255) & ~(size_t)255; return r;
  };
  float*  h      = (float*)alloc((size_t)NN * HH * 4);
  float*  agg    = (float*)alloc((size_t)NN * HH * 4);
  float*  pooled = (float*)alloc((size_t)GG * HH * 4);
  __bf16* h_bf   = (__bf16*)alloc((size_t)NN * HH * 2);
  __bf16* e_bf   = (__bf16*)alloc((size_t)EE * HH * 2);
  __bf16* wbuf   = (__bf16*)alloc((size_t)LL * WBLK_ELEMS * 2);

  // Weight prep: bf16 + transpose to column-major, packed per layer.
  for (int l = 0; l < LL; ++l) {
    __bf16* base = wbuf + (size_t)l * WBLK_ELEMS;
    transpose_cvt<<<(HH * HH + 255) / 256, 256, 0, stream>>>(
        em_w1 + (size_t)l * HH * HH, base, HH, HH);
    transpose_cvt<<<(HH * HH + 255) / 256, 256, 0, stream>>>(
        em_w2 + (size_t)l * HH * HH, base + 16384, HH, HH);
    transpose_cvt<<<(2 * HH * HH + 255) / 256, 256, 0, stream>>>(
        nm_w1 + (size_t)l * 2 * HH * HH, base + 32768, 2 * HH, HH);
    transpose_cvt<<<(HH * HH + 255) / 256, 256, 0, stream>>>(
        nm_w2 + (size_t)l * HH * HH, base + 65536, HH, HH);
  }

  node_embed<<<NN, HH, 0, stream>>>(x, ne_w, ne_b, h, h_bf, agg);
  edge_embed<<<EE, HH, 0, stream>>>(ea, ee_w, ee_b, e_bf);

  const size_t shmem = (size_t)(WBLK_ELEMS + 8 * STAGE_ELEMS) * 2;  // 229376 B
  for (int l = 0; l < LL; ++l) {
    fused_edge_layer<<<1024, 256, shmem, stream>>>(
        srcI, dstI, h_bf, e_bf, agg, wbuf + (size_t)l * WBLK_ELEMS,
        em_b1 + l * HH, em_b2 + l * HH, nm_b1 + l * HH, nm_b2 + l * HH,
        bn_g + l * HH, bn_b + l * HH);
    update_h<<<(NN * HH + 255) / 256, 256, 0, stream>>>(h, agg, h_bf, NN * HH);
  }

  zero_f32<<<(GG * HH + 255) / 256, 256, 0, stream>>>(pooled, GG * HH);
  pool_scatter<<<(NN * HH + 255) / 256, 256, 0, stream>>>(h, batch, pooled, NN * HH);
  readout<<<GG, HH, 0, stream>>>(pooled, ro_w1, ro_b1, ro_w2, ro_b2, (float*)d_out);
}
